// Head_13477607375352
// MI455X (gfx1250) — matrix operand order repaired and checked by reading, not compiled, not run on hardware
//
#include <hip/hip_runtime.h>
#include <hip/hip_bf16.h>

#define EMBD  1024
#define HEAD  64
#define SEQ   4096
#define BATCH 4
#define BK    64          // keys per flash iteration

typedef __bf16 v16bf __attribute__((ext_vector_type(16)));
typedef float  v8f   __attribute__((ext_vector_type(8)));

// ---------------------------------------------------------------------------
// Kernel 1: QKV projection.  [T,C] x [C,H] -> bf16 [B,T,H] per matrix.
// One wave computes one 16x16 output tile via v_wmma_f32_16x16x32_bf16,
// K-loop over C=1024 in steps of 32 (32 WMMA per tile).
// grid = (T/16, 3, B), block = 128 (4 waves, each wave = one 16-col slice of H)
// ---------------------------------------------------------------------------
__global__ __launch_bounds__(128)
void qkv_proj_kernel(const float* __restrict__ x,
                     const float* __restrict__ Wq,
                     const float* __restrict__ Wk,
                     const float* __restrict__ Wv,
                     __bf16* __restrict__ qkv_ws)   // [3][B][T][H] bf16
{
    const int wave = threadIdx.x >> 5;
    const int lane = threadIdx.x & 31;
    const int t0   = blockIdx.x * 16;
    const int mat  = blockIdx.y;          // 0=Q 1=K 2=V
    const int b    = blockIdx.z;

    const float* W  = (mat == 0) ? Wq : (mat == 1) ? Wk : Wv;
    const int    h0 = wave * 16;

    // ISA 16-bit A layout: row = lane&15; lanes<16 cover K {0..7,16..23},
    // lanes>=16 cover K {8..15,24..31}.
    const int row = lane & 15;
    const int kba = (lane < 16) ? 0 : 8;    // A-matrix K base
    const int kbb = (lane < 16) ? 0 : 16;   // unused (B operand uses kba)
    const int col = lane & 15;              // B/D column = lane

    const float* xrow = x + ((size_t)b * SEQ + (size_t)(t0 + row)) * EMBD;

    v8f acc = {};
    for (int c0 = 0; c0 < EMBD; c0 += 32) {
        // A operand: 16x32 tile of x, converted f32 -> bf16
        v16bf a;
        const float* p0 = xrow + c0 + kba;        // K = kba + 0..7
        const float* p1 = xrow + c0 + kba + 16;   // K = kba + 16..23
#pragma unroll
        for (int e = 0; e < 8; ++e) {
            a[e]     = (__bf16)p0[e];
            a[e + 8] = (__bf16)p1[e];
        }
        // B operand: 32x16 tile of W (K x N), element (k,n) = W[(c0+k)*H + h0+n]
        v16bf bm;
#pragma unroll
        for (int e = 0; e < 16; ++e)
            bm[e] = (__bf16)W[(size_t)(c0 + kba + (e & 7) + ((e >> 3) << 4)) * HEAD + h0 + col];

        acc = __builtin_amdgcn_wmma_f32_16x16x32_bf16(
                  false, a, false, bm, (short)0, acc, false, false);
    }

    // fold softmax scale 1/sqrt(64) into Q
    const float scl = (mat == 0) ? 0.125f : 1.0f;
    __bf16* out = qkv_ws + (size_t)mat * BATCH * SEQ * HEAD
                         + (size_t)b * SEQ * HEAD;
#pragma unroll
    for (int r = 0; r < 8; ++r) {
        const int m = t0 + r + ((lane >> 4) << 3);   // D layout: rows r / r+8
        out[(size_t)m * HEAD + h0 + col] = (__bf16)(acc[r] * scl);
    }
}

// ---------------------------------------------------------------------------
// Kernel 2: flash attention (online softmax), bf16 WMMA, f32 accum.
// WG = 4 waves x 16 Q-rows = 64 rows.  Key loop streams BK=64 keys/iter;
// K/V tiles staged into LDS with GLOBAL_LOAD_ASYNC_TO_LDS_B128 (ASYNCcnt),
// shared by all 4 waves.  16 WMMA per iteration per wave.
// grid = (T/64, B), block = 128
// ---------------------------------------------------------------------------
__global__ __launch_bounds__(128)
void flash_attn_kernel(const __bf16* __restrict__ Q,
                       const __bf16* __restrict__ K,
                       const __bf16* __restrict__ V,
                       float* __restrict__ out)     // [B,T,H] f32
{
    __shared__ __align__(16) __bf16 Ks[BK][HEAD];     // 8 KB
    __shared__ __align__(16) __bf16 Vs[BK][HEAD];     // 8 KB
    __shared__ __align__(16) __bf16 Ps[4][16][BK];    // 8 KB (per-wave P staging)

    const int wave = threadIdx.x >> 5;
    const int lane = threadIdx.x & 31;
    const int b    = blockIdx.y;
    const int m0   = blockIdx.x * 64 + wave * 16;

    const int row = lane & 15;
    const int kba = (lane < 16) ? 0 : 8;
    const int kbb = (lane < 16) ? 0 : 16;
    const int col = lane & 15;

    const __bf16* Qb = Q + (size_t)b * SEQ * HEAD;
    const __bf16* Kb = K + (size_t)b * SEQ * HEAD;
    const __bf16* Vb = V + (size_t)b * SEQ * HEAD;

    // Load this wave's Q rows once, A-layout, two 16x32 chunks over H=64
    v16bf qa[2];
#pragma unroll
    for (int ck = 0; ck < 2; ++ck) {
        const __bf16* qrow = Qb + (size_t)(m0 + row) * HEAD + 32 * ck;
#pragma unroll
        for (int e = 0; e < 8; ++e) {
            qa[ck][e]     = qrow[kba + e];
            qa[ck][e + 8] = qrow[kba + 16 + e];
        }
    }

    // LDS byte offsets (generic LDS addresses carry the aperture tag in the
    // high 32 bits only, so truncation yields the raw LDS offset).
    const unsigned ldsK0 = (unsigned)(uintptr_t)&Ks[0][0];
    const unsigned ldsV0 = (unsigned)(uintptr_t)&Vs[0][0];

    v8f Ot[4] = {{}, {}, {}, {}};       // O accumulator [16 x 64] f32
    float mrow[8], lrow[8];
#pragma unroll
    for (int r = 0; r < 8; ++r) { mrow[r] = -3.0e38f; lrow[r] = 0.0f; }

    for (int s0 = 0; s0 < SEQ; s0 += BK) {
        // ---- async stage of K/V tiles (BK x 64 bf16 = 8 KB = 512 x b128 each)
        {
            const char* gk = (const char*)(Kb + (size_t)s0 * HEAD);
            const char* gv = (const char*)(Vb + (size_t)s0 * HEAD);
#pragma unroll
            for (int i = 0; i < 4; ++i) {
                const unsigned byteoff = (threadIdx.x + i * 128) * 16u;
                const unsigned lk = ldsK0 + byteoff;
                const unsigned lv = ldsV0 + byteoff;
                const char* ga = gk + byteoff;
                const char* gb = gv + byteoff;
                asm volatile("global_load_async_to_lds_b128 %0, %1, off"
                             :: "v"(lk), "v"(ga) : "memory");
                asm volatile("global_load_async_to_lds_b128 %0, %1, off"
                             :: "v"(lv), "v"(gb) : "memory");
            }
            if (s0 + BK < SEQ) {   // uniform branch; EXEC stays all-ones
                __builtin_prefetch(Kb + (size_t)(s0 + BK) * HEAD + lane * 64, 0, 0);
                __builtin_prefetch(Vb + (size_t)(s0 + BK) * HEAD + lane * 64, 0, 0);
            }
            asm volatile("s_wait_asynccnt 0x0" ::: "memory");
        }
        __syncthreads();

        // ---- scores S = Q * K^T  (four 16x16 n-tiles over BK=64 keys)
        v8f S[4] = {{}, {}, {}, {}};
#pragma unroll
        for (int t = 0; t < 4; ++t) {
#pragma unroll
            for (int ck = 0; ck < 2; ++ck) {
                v16bf bk;
#pragma unroll
                for (int e = 0; e < 16; ++e)
                    bk[e] = Ks[t * 16 + col][32 * ck + kba + (e & 7) + ((e >> 3) << 4)];
                S[t] = __builtin_amdgcn_wmma_f32_16x16x32_bf16(
                           false, qa[ck], false, bk, (short)0, S[t], false, false);
            }
        }

        // ---- online softmax (per row; butterfly over the 16-lane group)
        float corr[8];
#pragma unroll
        for (int r = 0; r < 8; ++r) {
            float mx = fmaxf(fmaxf(S[0][r], S[1][r]), fmaxf(S[2][r], S[3][r]));
#pragma unroll
            for (int off = 8; off >= 1; off >>= 1)
                mx = fmaxf(mx, __shfl_xor(mx, off, 32));
            const float nm = fmaxf(mrow[r], mx);
            corr[r] = __expf(mrow[r] - nm);
            mrow[r] = nm;
            float rs = 0.0f;
#pragma unroll
            for (int t = 0; t < 4; ++t) {
                const float p = __expf(S[t][r] - nm);
                S[t][r] = p;
                rs += p;
            }
#pragma unroll
            for (int off = 8; off >= 1; off >>= 1)
                rs += __shfl_xor(rs, off, 32);
            lrow[r] = lrow[r] * corr[r] + rs;
        }
#pragma unroll
        for (int r = 0; r < 8; ++r) {
            Ot[0][r] *= corr[r]; Ot[1][r] *= corr[r];
            Ot[2][r] *= corr[r]; Ot[3][r] *= corr[r];
        }

        // ---- D-layout -> A-layout transpose of P through per-wave LDS
#pragma unroll
        for (int r = 0; r < 8; ++r) {
            const int m = r + ((lane >> 4) << 3);
#pragma unroll
            for (int t = 0; t < 4; ++t)
                Ps[wave][m][t * 16 + col] = (__bf16)S[t][r];
        }
        asm volatile("s_wait_dscnt 0x0" ::: "memory");   // wave-local LDS RAW
        v16bf pa[2];                                     // P as A: 16 x 64 keys
#pragma unroll
        for (int kc = 0; kc < 2; ++kc) {
#pragma unroll
            for (int e = 0; e < 8; ++e) {
                pa[kc][e]     = Ps[wave][row][kc * 32 + kba + e];
                pa[kc][e + 8] = Ps[wave][row][kc * 32 + kba + 16 + e];
            }
        }

        // ---- O += P * V   (K = 64 keys in two 32-chunks, N = 64 in four tiles)
#pragma unroll
        for (int nt = 0; nt < 4; ++nt) {
#pragma unroll
            for (int kc = 0; kc < 2; ++kc) {
                v16bf bv;
#pragma unroll
                for (int e = 0; e < 16; ++e)
                    bv[e] = Vs[kc * 32 + kba + (e & 7) + ((e >> 3) << 4)][nt * 16 + col];
                Ot[nt] = __builtin_amdgcn_wmma_f32_16x16x32_bf16(
                             false, pa[kc], false, bv, (short)0, Ot[nt], false, false);
            }
        }
        __syncthreads();   // protect Ks/Vs before next stage
    }

    // ---- epilogue: normalize and write f32 output
#pragma unroll
    for (int nt = 0; nt < 4; ++nt) {
#pragma unroll
        for (int r = 0; r < 8; ++r) {
            const int m = m0 + r + ((lane >> 4) << 3);
            out[((size_t)b * SEQ + m) * HEAD + nt * 16 + col] = Ot[nt][r] / lrow[r];
        }
    }
}

// ---------------------------------------------------------------------------
extern "C" void kernel_launch(void* const* d_in, const int* in_sizes, int n_in,
                              void* d_out, int out_size, void* d_ws, size_t ws_size,
                              hipStream_t stream) {
    const float* x  = (const float*)d_in[0];
    const float* Wq = (const float*)d_in[1];
    const float* Wk = (const float*)d_in[2];
    const float* Wv = (const float*)d_in[3];

    __bf16* ws = (__bf16*)d_ws;                       // 3*B*T*H bf16 = 6 MB
    __bf16* Qw = ws;
    __bf16* Kw = ws + (size_t)1 * BATCH * SEQ * HEAD;
    __bf16* Vw = ws + (size_t)2 * BATCH * SEQ * HEAD;

    qkv_proj_kernel<<<dim3(SEQ / 16, 3, BATCH), 128, 0, stream>>>(x, Wq, Wk, Wv, ws);
    flash_attn_kernel<<<dim3(SEQ / 64, BATCH), 128, 0, stream>>>(Qw, Kw, Vw, (float*)d_out);
}
